// Node_Edge_cls_Module_75788992905506
// MI455X (gfx1250) — matrix operand order repaired and checked by reading, not compiled, not run on hardware
//
#include <hip/hip_runtime.h>
#include <hip/hip_bf16.h>

// ---------------------------------------------------------------------------
// MI455X (gfx1250, wave32) implementation.
// All dense GEMMs use v_wmma_f32_16x16x32_f16 (f16 in, f32 acc).
// Attention is fused (flash-style) with WMMA: S=QK^T is one 16x16x32 WMMA per
// key tile (dh==32==WMMA K), P@V is two WMMAs with P zero-padded to K=32.
// Data movement: interior tiles use vector (float2/float4) global loads,
// packed f32->f16 converts, and 16-byte aligned LDS accesses so fragments are
// ds_load_b128 pairs instead of scalar u16 gathers.
// ---------------------------------------------------------------------------

typedef _Float16 f16;
typedef _Float16 v16h __attribute__((ext_vector_type(16)));
typedef _Float16 h8v  __attribute__((ext_vector_type(8)));
typedef float    v8f  __attribute__((ext_vector_type(8)));

#define NG 8      // graphs (batch)
#define NN 1024   // nodes per graph
#define NE 8192   // edges per graph
#define NT (NE+NN)
#define FIN 64
#define CC 128    // GAT channels per head
#define HGC 256   // 2 heads * 128
#define DD 128    // model dim
#define HA 4      // attention heads
#define DH 32     // head dim
#define CDIV(a,b) (((a)+(b)-1)/(b))

// output layout (floats)
#define O_NP   0            // [8192,7]
#define O_EP   57344        // [65536,2]
#define O_VAD  188416       // [8192,2]
#define O_BND  204800       // [8,5,2]
#define O_MSK  204880       // [8,5]

union AF { v16h v; f16 h[16]; };
union CF { v8f  v; float f[8]; };

__device__ inline void atomicMaxF(float* addr, float val) {
  int* ia = (int*)addr;
  int old = *ia;
  while (__int_as_float(old) < val) {
    int assumed = old;
    old = atomicCAS(ia, assumed, __float_as_int(val));
    if (old == assumed) break;
  }
}

// ---------------------------------------------------------------------------
// WMMA GEMM: C[M,N] = act(A[M,K] * op(B) + bias), f32 I/O, f16 compute.
// transB==0: B is [K,N] row-major.  transB==1: B is [N,K] row-major (B^T).
// 128 threads = 4 waves, block tile 32x32, K step 32.
// LDS: As[r][k] and BsT[n][k], both k-contiguous with 80B row stride so each
// wmma fragment is two aligned 16-byte ds_load_b128s.
// ---------------------------------------------------------------------------
__global__ __launch_bounds__(128)
void gemm_wmma(const float* __restrict__ A, const float* __restrict__ Bm,
               const float* __restrict__ bias, float* __restrict__ C,
               int M, int N, int K, int transB, int relu)
{
  __shared__ f16 As [32][40];
  __shared__ f16 BsT[32][40];
  const int tid  = threadIdx.x;
  const int lane = tid & 31;
  const int wv   = tid >> 5;
  const int wr   = (wv >> 1) << 4;     // wave row offset in block tile
  const int wc   = (wv & 1) << 4;      // wave col offset
  const int row0 = blockIdx.y * 32;
  const int col0 = blockIdx.x * 32;
  const int m    = lane & 15;
  const int half = lane >> 4;

  CF acc;
#pragma unroll
  for (int i = 0; i < 8; i++) acc.f[i] = 0.f;

  for (int k0 = 0; k0 < K; k0 += 32) {
    const bool kfull = (k0 + 32 <= K);

    // ---- stage A tile: As[r][kk..kk+7], f32 -> f16 ----
    {
      const int r = tid >> 2, kk = (tid & 3) * 8;
      const size_t base = (size_t)(row0 + r) * K + k0 + kk;
      const bool rok = (row0 + r) < M;
      if (kfull && ((K & 1) == 0)) {
        float2 p0{0.f,0.f}, p1{0.f,0.f}, p2{0.f,0.f}, p3{0.f,0.f};
        if (rok) {
          p0 = *(const float2*)(A + base + 0);
          p1 = *(const float2*)(A + base + 2);
          p2 = *(const float2*)(A + base + 4);
          p3 = *(const float2*)(A + base + 6);
        }
        h8v pk;
        pk[0]=(f16)p0.x; pk[1]=(f16)p0.y; pk[2]=(f16)p1.x; pk[3]=(f16)p1.y;
        pk[4]=(f16)p2.x; pk[5]=(f16)p2.y; pk[6]=(f16)p3.x; pk[7]=(f16)p3.y;
        *(h8v*)&As[r][kk] = pk;
      } else {
#pragma unroll
        for (int j = 0; j < 8; j++) {
          float v = 0.f;
          if (rok && (k0 + kk + j) < K) v = A[base + j];
          As[r][kk + j] = (f16)v;
        }
      }
    }

    // ---- stage B tile as BsT[n][k] ----
    if (transB) {
      const int n = tid >> 2, kk = (tid & 3) * 8;
      const size_t base = (size_t)(col0 + n) * K + k0 + kk;
      const bool nok = (col0 + n) < N;
      if (kfull && ((K & 1) == 0)) {
        float2 p0{0.f,0.f}, p1{0.f,0.f}, p2{0.f,0.f}, p3{0.f,0.f};
        if (nok) {
          p0 = *(const float2*)(Bm + base + 0);
          p1 = *(const float2*)(Bm + base + 2);
          p2 = *(const float2*)(Bm + base + 4);
          p3 = *(const float2*)(Bm + base + 6);
        }
        h8v pk;
        pk[0]=(f16)p0.x; pk[1]=(f16)p0.y; pk[2]=(f16)p1.x; pk[3]=(f16)p1.y;
        pk[4]=(f16)p2.x; pk[5]=(f16)p2.y; pk[6]=(f16)p3.x; pk[7]=(f16)p3.y;
        *(h8v*)&BsT[n][kk] = pk;
      } else {
#pragma unroll
        for (int j = 0; j < 8; j++) {
          float v = 0.f;
          if (nok && (k0 + kk + j) < K) v = Bm[base + j];
          BsT[n][kk + j] = (f16)v;
        }
      }
    } else {
      const int kx = tid >> 2, n0 = (tid & 3) * 8;
      const size_t base = (size_t)(k0 + kx) * N + col0 + n0;
      const bool kok = (k0 + kx) < K;
      if (kfull && (col0 + 32 <= N) && ((N & 1) == 0)) {
        float2 p0 = *(const float2*)(Bm + base + 0);
        float2 p1 = *(const float2*)(Bm + base + 2);
        float2 p2 = *(const float2*)(Bm + base + 4);
        float2 p3 = *(const float2*)(Bm + base + 6);
        BsT[n0 + 0][kx] = (f16)p0.x; BsT[n0 + 1][kx] = (f16)p0.y;
        BsT[n0 + 2][kx] = (f16)p1.x; BsT[n0 + 3][kx] = (f16)p1.y;
        BsT[n0 + 4][kx] = (f16)p2.x; BsT[n0 + 5][kx] = (f16)p2.y;
        BsT[n0 + 6][kx] = (f16)p3.x; BsT[n0 + 7][kx] = (f16)p3.y;
      } else {
#pragma unroll
        for (int j = 0; j < 8; j++) {
          float v = 0.f;
          if (kok && (col0 + n0 + j) < N) v = Bm[base + j];
          BsT[n0 + j][kx] = (f16)v;
        }
      }
    }

    // prefetch next K tile (global_prefetch_b8)
    if (k0 + 32 < K) {
      if (row0 + (tid >> 2) < M)
        __builtin_prefetch(&A[(size_t)(row0 + (tid >> 2)) * K + k0 + 32], 0, 0);
    }
    __syncthreads();

    // fragments: two aligned 16B LDS loads each
    AF af, bf;
    {
      const h8v alo = *(const h8v*)&As[wr + m][half * 8];
      const h8v ahi = *(const h8v*)&As[wr + m][16 + half * 8];
#pragma unroll
      for (int i = 0; i < 8; i++) { af.h[i] = alo[i]; af.h[8 + i] = ahi[i]; }
      const h8v blo = *(const h8v*)&BsT[wc + m][half * 8];
      const h8v bhi = *(const h8v*)&BsT[wc + m][16 + half * 8];
#pragma unroll
      for (int i = 0; i < 8; i++) { bf.h[i] = blo[i]; bf.h[8 + i] = bhi[i]; }
    }

    acc.v = __builtin_amdgcn_wmma_f32_16x16x32_f16(false, af.v, false, bf.v,
                                                   (short)0, acc.v, false, false);
    __syncthreads();
  }

#pragma unroll
  for (int v2 = 0; v2 < 8; v2++) {
    int r = row0 + wr + half * 8 + v2;
    int c = col0 + wc + m;
    if (r < M && c < N) {
      float val = acc.f[v2];
      if (bias) val += bias[c];
      if (relu) val = fmaxf(val, 0.f);
      C[(size_t)r * N + c] = val;
    }
  }
}

// ---------------------------------------------------------------------------
// Fused flash MHA for the first mha(x,x,x): one wave per (16-query tile, head,
// graph). qkv layout [b][n][384] = [q|k|v] each 4 heads * 32.
// Writes obuf[b][n][128] (heads concatenated).
// ---------------------------------------------------------------------------
__global__ __launch_bounds__(32)
void flash_mha(const float* __restrict__ qkv, float* __restrict__ obuf)
{
  const int lane = threadIdx.x;
  const int qt = blockIdx.x, h = blockIdx.y, b = blockIdx.z;
  const int m = lane & 15, half = lane >> 4;

  __shared__ float Sld[16][17];
  __shared__ f16   Ps[16][40];     // P zero-padded to K=32, 80B row stride
  __shared__ f16   VsT[32][24];    // V tile transposed: VsT[dh][key], 48B rows
  __shared__ float cscale[16];
  __shared__ float linv[16];

  for (int i = lane; i < 16 * 40; i += 32) ((f16*)Ps)[i] = (f16)0.f;

  const float scl = 0.17677669529663687f;  // 1/sqrt(32)
  // Q fragment: two contiguous 8-float runs -> 4 aligned float4 loads
  AF qa;
  {
    const float* qrow = &qkv[((size_t)b * NN + qt * 16 + m) * 384 + h * 32];
    float4 q0 = *(const float4*)(qrow + half * 8);
    float4 q1 = *(const float4*)(qrow + half * 8 + 4);
    float4 q2 = *(const float4*)(qrow + 16 + half * 8);
    float4 q3 = *(const float4*)(qrow + 16 + half * 8 + 4);
    qa.h[0]  = (f16)(scl * q0.x); qa.h[1]  = (f16)(scl * q0.y);
    qa.h[2]  = (f16)(scl * q0.z); qa.h[3]  = (f16)(scl * q0.w);
    qa.h[4]  = (f16)(scl * q1.x); qa.h[5]  = (f16)(scl * q1.y);
    qa.h[6]  = (f16)(scl * q1.z); qa.h[7]  = (f16)(scl * q1.w);
    qa.h[8]  = (f16)(scl * q2.x); qa.h[9]  = (f16)(scl * q2.y);
    qa.h[10] = (f16)(scl * q2.z); qa.h[11] = (f16)(scl * q2.w);
    qa.h[12] = (f16)(scl * q3.x); qa.h[13] = (f16)(scl * q3.y);
    qa.h[14] = (f16)(scl * q3.z); qa.h[15] = (f16)(scl * q3.w);
  }
  CF o0, o1;
#pragma unroll
  for (int i = 0; i < 8; i++) { o0.f[i] = 0.f; o1.f[i] = 0.f; }
  float mrun = -1e30f, lrun = 0.f;   // valid in lanes 0..15 (row = lane)
  __syncthreads();

  for (int kt = 0; kt < NN / 16; kt++) {
    // B frag for S=Q*K^T: Bmat[kdh][key]=K[key][kdh]; two contiguous 8-float runs
    AF bf;
    {
      const float* krow = &qkv[((size_t)b * NN + kt * 16 + m) * 384 + 128 +
                               h * 32 + half * 8];
      float4 k0 = *(const float4*)(krow + 0);
      float4 k1 = *(const float4*)(krow + 4);
      float4 k2 = *(const float4*)(krow + 16);
      float4 k3 = *(const float4*)(krow + 20);
      bf.h[0]  = (f16)k0.x; bf.h[1]  = (f16)k0.y; bf.h[2]  = (f16)k0.z; bf.h[3]  = (f16)k0.w;
      bf.h[4]  = (f16)k1.x; bf.h[5]  = (f16)k1.y; bf.h[6]  = (f16)k1.z; bf.h[7]  = (f16)k1.w;
      bf.h[8]  = (f16)k2.x; bf.h[9]  = (f16)k2.y; bf.h[10] = (f16)k2.z; bf.h[11] = (f16)k2.w;
      bf.h[12] = (f16)k3.x; bf.h[13] = (f16)k3.y; bf.h[14] = (f16)k3.z; bf.h[15] = (f16)k3.w;
    }
    // stage V tile transposed (coalesced float4 global loads, b16 LDS scatter)
    {
      const float* vrow = &qkv[((size_t)b * NN + kt * 16 + m) * 384 + 256 +
                               h * 32 + half * 16];
      float4 u0 = *(const float4*)(vrow + 0);
      float4 u1 = *(const float4*)(vrow + 4);
      float4 u2 = *(const float4*)(vrow + 8);
      float4 u3 = *(const float4*)(vrow + 12);
      const int d0 = half * 16;
      VsT[d0 +  0][m] = (f16)u0.x; VsT[d0 +  1][m] = (f16)u0.y;
      VsT[d0 +  2][m] = (f16)u0.z; VsT[d0 +  3][m] = (f16)u0.w;
      VsT[d0 +  4][m] = (f16)u1.x; VsT[d0 +  5][m] = (f16)u1.y;
      VsT[d0 +  6][m] = (f16)u1.z; VsT[d0 +  7][m] = (f16)u1.w;
      VsT[d0 +  8][m] = (f16)u2.x; VsT[d0 +  9][m] = (f16)u2.y;
      VsT[d0 + 10][m] = (f16)u2.z; VsT[d0 + 11][m] = (f16)u2.w;
      VsT[d0 + 12][m] = (f16)u3.x; VsT[d0 + 13][m] = (f16)u3.y;
      VsT[d0 + 14][m] = (f16)u3.z; VsT[d0 + 15][m] = (f16)u3.w;
    }

    CF sf;
#pragma unroll
    for (int i = 0; i < 8; i++) sf.f[i] = 0.f;
    sf.v = __builtin_amdgcn_wmma_f32_16x16x32_f16(false, qa.v, false, bf.v,
                                                  (short)0, sf.v, false, false);
#pragma unroll
    for (int v2 = 0; v2 < 8; v2++) Sld[half * 8 + v2][m] = sf.f[v2];
    __syncthreads();

    if (lane < 16) {   // online softmax, one row per lane
      float rm = -1e30f;
#pragma unroll
      for (int j = 0; j < 16; j++) rm = fmaxf(rm, Sld[lane][j]);
      float nm = fmaxf(mrun, rm);
      float corr = __expf(mrun - nm);
      float ssum = 0.f;
#pragma unroll
      for (int j = 0; j < 16; j++) {
        float p = __expf(Sld[lane][j] - nm);
        Ps[lane][j] = (f16)p;
        ssum += p;
      }
      lrun = lrun * corr + ssum;
      mrun = nm;
      cscale[lane] = corr;
    }
    __syncthreads();

#pragma unroll
    for (int v2 = 0; v2 < 8; v2++) {
      float cc = cscale[half * 8 + v2];
      o0.f[v2] *= cc; o1.f[v2] *= cc;
    }

    // P fragment: two aligned 16B LDS loads
    AF pa;
    {
      const h8v plo = *(const h8v*)&Ps[m][half * 16];
      const h8v phi = *(const h8v*)&Ps[m][half * 16 + 8];
#pragma unroll
      for (int i = 0; i < 8; i++) { pa.h[i] = plo[i]; pa.h[8 + i] = phi[i]; }
    }
    // V fragments: half==1 lanes are the zero-padded K rows by construction
    AF vb0, vb1;
#pragma unroll
    for (int i = 0; i < 16; i++) { vb0.h[i] = (f16)0.f; vb1.h[i] = (f16)0.f; }
    if (half == 0) {
      const h8v a0 = *(const h8v*)&VsT[m][0];
      const h8v a1 = *(const h8v*)&VsT[m][8];
      const h8v c0 = *(const h8v*)&VsT[16 + m][0];
      const h8v c1 = *(const h8v*)&VsT[16 + m][8];
#pragma unroll
      for (int i = 0; i < 8; i++) {
        vb0.h[i] = a0[i]; vb0.h[8 + i] = a1[i];
        vb1.h[i] = c0[i]; vb1.h[8 + i] = c1[i];
      }
    }
    o0.v = __builtin_amdgcn_wmma_f32_16x16x32_f16(false, pa.v, false, vb0.v,
                                                  (short)0, o0.v, false, false);
    o1.v = __builtin_amdgcn_wmma_f32_16x16x32_f16(false, pa.v, false, vb1.v,
                                                  (short)0, o1.v, false, false);
    __syncthreads();
  }

  if (lane < 16) linv[lane] = 1.f / (lrun + 1e-16f);
  __syncthreads();
#pragma unroll
  for (int v2 = 0; v2 < 8; v2++) {
    int rl = half * 8 + v2;
    int row = qt * 16 + rl;
    float li = linv[rl];
    obuf[((size_t)b * NN + row) * DD + h * 32 + 0  + m] = o0.f[v2] * li;
    obuf[((size_t)b * NN + row) * DD + h * 32 + 16 + m] = o1.f[v2] * li;
  }
}

// --------------------------- small kernels ---------------------------------

__global__ void fill_f32(float* p, float v, int n) {
  int i = blockIdx.x * blockDim.x + threadIdx.x;
  if (i < n) p[i] = v;
}

__global__ void edge_accum(const float* __restrict__ ea, const int* __restrict__ dst,
                           float* deg, float* easum) {
  int idx = blockIdx.x * blockDim.x + threadIdx.x;
  if (idx >= NG * NE) return;
  int b = idx / NE;
  int d = dst[idx];
  atomicAdd(&deg[b * NN + d], 1.f);
  for (int k = 0; k < 16; k++)
    atomicAdd(&easum[(size_t)(b * NN + d) * 16 + k], ea[(size_t)idx * 16 + k]);
}

__global__ void build_ea2(const float* __restrict__ ea, const float* __restrict__ easum,
                          const float* __restrict__ deg, float* __restrict__ ea2) {
  int idx = blockIdx.x * blockDim.x + threadIdx.x;
  if (idx >= NG * NT * 16) return;
  int k = idx & 15;
  int i = (idx >> 4) % NT;
  int b = idx / (NT * 16);
  float v;
  if (i < NE) v = ea[((size_t)b * NE + i) * 16 + k];
  else        v = easum[((size_t)b * NN + (i - NE)) * 16 + k] / fmaxf(deg[b * NN + (i - NE)], 1.f);
  ea2[idx] = v;
}

// wae[k][hh] = sum_c We[k, hh*128+c] * att_edge[hh,c]   (16x2)
__global__ void wae_kernel(const float* __restrict__ We, const float* __restrict__ attE,
                           float* __restrict__ wae) {
  int t = threadIdx.x;
  if (t >= 32) return;
  int k = t >> 1, hh = t & 1;
  float s = 0.f;
  for (int c = 0; c < CC; c++) s += We[(size_t)k * HGC + hh * CC + c] * attE[hh * CC + c];
  wae[k * 2 + hh] = s;
}

__global__ void ae_kernel(const float* __restrict__ ea2, const float* __restrict__ wae,
                          float* __restrict__ a_e) {
  int idx = blockIdx.x * blockDim.x + threadIdx.x;
  if (idx >= NG * NT * 2) return;
  int hh = idx & 1;
  int row = idx >> 1;           // b*NT + i
  float s = 0.f;
  for (int k = 0; k < 16; k++) s += ea2[(size_t)row * 16 + k] * wae[k * 2 + hh];
  a_e[idx] = s;
}

__global__ void asrc_adst(const float* __restrict__ h, const float* __restrict__ attS,
                          const float* __restrict__ attD, float* a_src, float* a_dst) {
  int idx = blockIdx.x * blockDim.x + threadIdx.x;
  if (idx >= NG * NN * 2) return;
  int hh = idx & 1;
  int row = idx >> 1;           // b*NN + n
  float s = 0.f, d = 0.f;
  for (int c = 0; c < CC; c++) {
    float hv = h[(size_t)row * HGC + hh * CC + c];
    s += hv * attS[hh * CC + c];
    d += hv * attD[hh * CC + c];
  }
  a_src[idx] = s; a_dst[idx] = d;
}

__global__ void alpha_max(const float* __restrict__ a_src, const float* __restrict__ a_dst,
                          const float* __restrict__ a_e, const int* __restrict__ src,
                          const int* __restrict__ dst, float* wbuf, float* segm) {
  int idx = blockIdx.x * blockDim.x + threadIdx.x;
  if (idx >= NG * NT * 2) return;
  int hh = idx & 1;
  int i = (idx >> 1) % NT;
  int b = idx / (NT * 2);
  int s, d;
  if (i < NE) { s = src[b * NE + i]; d = dst[b * NE + i]; }
  else        { s = i - NE; d = i - NE; }
  float al = a_src[(b * NN + s) * 2 + hh] + a_dst[(b * NN + d) * 2 + hh] + a_e[idx];
  al = (al > 0.f) ? al : 0.2f * al;     // leaky_relu 0.2
  wbuf[idx] = al;
  atomicMaxF(&segm[(b * NN + d) * 2 + hh], al);
}

__global__ void expsum(const int* __restrict__ src, const int* __restrict__ dst,
                       const float* __restrict__ segm, float* wbuf, float* segs) {
  int idx = blockIdx.x * blockDim.x + threadIdx.x;
  if (idx >= NG * NT * 2) return;
  int hh = idx & 1;
  int i = (idx >> 1) % NT;
  int b = idx / (NT * 2);
  int d = (i < NE) ? dst[b * NE + i] : (i - NE);
  float w = __expf(wbuf[idx] - segm[(b * NN + d) * 2 + hh]);
  wbuf[idx] = w;
  atomicAdd(&segs[(b * NN + d) * 2 + hh], w);
}

__global__ void aggregate(const float* __restrict__ h, const int* __restrict__ src,
                          const int* __restrict__ dst, const float* __restrict__ wbuf,
                          const float* __restrict__ segs, float* gacc) {
  int idx = blockIdx.x * blockDim.x + threadIdx.x;
  if (idx >= NG * NT * CC) return;
  int c = idx & (CC - 1);
  int rest = idx >> 7;
  int i = rest % NT;
  int b = rest / NT;
  int s, d;
  if (i < NE) { s = src[b * NE + i]; d = dst[b * NE + i]; }
  else        { s = i - NE; d = i - NE; }
  for (int hh = 0; hh < 2; hh++) {
    float coef = wbuf[((size_t)b * NT + i) * 2 + hh] /
                 (segs[(b * NN + d) * 2 + hh] + 1e-16f);
    atomicAdd(&gacc[((size_t)b * NN + d) * HGC + hh * CC + c],
              h[((size_t)b * NN + s) * HGC + hh * CC + c] * coef);
  }
}

__global__ void gat_final(const float* __restrict__ gacc, const float* __restrict__ bias,
                          float* __restrict__ out, int relu) {
  int idx = blockIdx.x * blockDim.x + threadIdx.x;
  if (idx >= NG * NN * CC) return;
  int c = idx & (CC - 1);
  int r = idx >> 7;
  float v = 0.5f * (gacc[(size_t)r * HGC + c] + gacc[(size_t)r * HGC + CC + c]) + bias[c];
  if (relu) v = fmaxf(v, 0.f);
  out[idx] = v;
}

__global__ void add_te(const float* __restrict__ emb, const float* __restrict__ te_w,
                       const float* __restrict__ te_b, float* __restrict__ xte) {
  int idx = blockIdx.x * blockDim.x + threadIdx.x;
  if (idx >= NG * NN * DD) return;
  int c = idx & (DD - 1);
  int n = (idx >> 7) % NN;
  float t = (float)n * (1.f / (NN - 1));
  xte[idx] = emb[idx] + t * te_w[c] + te_b[c];
}

__global__ __launch_bounds__(128)
void pooled_mask(const float* __restrict__ ao, const float* __restrict__ w1,
                 const float* __restrict__ b1, const float* __restrict__ w2,
                 const float* __restrict__ b2, float* __restrict__ mask_out) {
  int b = blockIdx.x, t = threadIdx.x;
  __shared__ float pooled[128], hid[128], lg5[8];
  float s = 0.f;
  for (int n = 0; n < NN; n++) s += ao[((size_t)b * NN + n) * DD + t];
  pooled[t] = s * (1.f / NN);
  __syncthreads();
  float hs = b1[t];
  for (int c = 0; c < DD; c++) hs += pooled[c] * w1[c * DD + t];
  hid[t] = fmaxf(hs, 0.f);
  __syncthreads();
  if (t < 5) {
    float s2 = b2[t];
    for (int c = 0; c < DD; c++) s2 += hid[c] * w2[c * 5 + t];
    lg5[t] = s2;
  }
  __syncthreads();
  if (t == 0) {
    int am = 0; float bm = lg5[0];
    for (int j = 1; j < 5; j++) if (lg5[j] > bm) { bm = lg5[j]; am = j; }
    for (int j = 0; j < 5; j++) mask_out[b * 5 + j] = (j <= am) ? 1.f : 0.f;
  }
}

__global__ void init_intervals(float* csb, float* ceb) {
  int i = threadIdx.x;
  if (i >= NG * 5) return;
  int k = i % 5;
  csb[i] = (float)k * 0.2f;
  ceb[i] = (float)(k + 1) * 0.2f;
}

__global__ void build_q(const float* __restrict__ ao, const float* __restrict__ csb,
                        const float* __restrict__ ceb, int ivx, float* __restrict__ q) {
  int idx = blockIdx.x * blockDim.x + threadIdx.x;
  if (idx >= NG * NN * 130) return;
  int col = idx % 130;
  int r = idx / 130;          // b*NN + n
  int b = r / NN;
  float v;
  if (col < DD)        v = ao[(size_t)r * DD + col];
  else if (col == 128) v = csb[b * 5 + ivx];
  else                 v = ceb[b * 5 + ivx];
  q[idx] = v;
}

__global__ void soeo_kernel(const float* __restrict__ lg, const float* __restrict__ wp,
                            float* __restrict__ so, float* __restrict__ eo) {
  int idx = blockIdx.x * blockDim.x + threadIdx.x;
  if (idx >= NG * NN) return;
  const float* p = &lg[(size_t)idx * 200];
  for (int part = 0; part < 2; part++) {
    const float* q = p + part * 100;
    float m = -1e30f;
    for (int j = 0; j < 100; j++) m = fmaxf(m, q[j]);
    float s = 0.f, acc = 0.f;
    for (int j = 0; j < 100; j++) {
      float e = __expf(q[j] - m);
      s += e; acc += e * wp[j];
    }
    float r = acc / s;
    if (part == 0) so[idx] = r; else eo[idx] = r;
  }
}

__global__ void rowstats(const float* __restrict__ qph, const float* __restrict__ kh2,
                         float* __restrict__ rsm, float* __restrict__ rsl) {
  int idx = blockIdx.x * blockDim.x + threadIdx.x;
  if (idx >= NG * HA * NN) return;
  int q = idx % NN;
  int h = (idx / NN) % HA;
  int b = idx / (NN * HA);
  const float* qv = &qph[((size_t)b * NN + q) * DD + h * DH];
  float mrun = -1e30f, l = 0.f;
  for (int k = 0; k < NN; k++) {
    const float* kv = &kh2[((size_t)b * NN + k) * DD + h * DH];
    float s = 0.f;
#pragma unroll
    for (int d = 0; d < DH; d++) s += qv[d] * kv[d];
    s *= 0.17677669529663687f;
    float nm = fmaxf(mrun, s);
    l = l * __expf(mrun - nm) + __expf(s - nm);
    mrun = nm;
  }
  rsm[idx] = mrun; rsl[idx] = l;
}

__global__ void colsum_acc(const float* __restrict__ qph, const float* __restrict__ kh2,
                           const float* __restrict__ rsm, const float* __restrict__ rsl,
                           float* __restrict__ colsum) {
  int idx = blockIdx.x * blockDim.x + threadIdx.x;
  if (idx >= NG * HA * NN) return;
  int q = idx % NN;
  int h = (idx / NN) % HA;
  int b = idx / (NN * HA);
  const float* qv = &qph[((size_t)b * NN + q) * DD + h * DH];
  float m = rsm[idx], linv = 0.25f / rsl[idx];   // /H for head mean
  for (int k = 0; k < NN; k++) {
    const float* kv = &kh2[((size_t)b * NN + k) * DD + h * DH];
    float s = 0.f;
#pragma unroll
    for (int d = 0; d < DH; d++) s += qv[d] * kv[d];
    s *= 0.17677669529663687f;
    atomicAdd(&colsum[b * NN + k], __expf(s - m) * linv);
  }
}

__global__ __launch_bounds__(256)
void update_intervals(const float* __restrict__ colsum, const float* __restrict__ so,
                      const float* __restrict__ eo, float* csb, float* ceb, int ivx) {
  int b = blockIdx.x, t = threadIdx.x;
  __shared__ float rs[256], re[256];
  float a = 0.f, c = 0.f;
  for (int k = t; k < NN; k += 256) {
    float w = colsum[b * NN + k];
    a += w * so[b * NN + k];
    c += w * eo[b * NN + k];
  }
  rs[t] = a; re[t] = c;
  __syncthreads();
  for (int off = 128; off > 0; off >>= 1) {
    if (t < off) { rs[t] += rs[t + off]; re[t] += re[t + off]; }
    __syncthreads();
  }
  if (t == 0) { csb[b * 5 + ivx] += rs[0]; ceb[b * 5 + ivx] += re[0]; }
}

__global__ void write_bounds(const float* csb, const float* ceb, float* ob) {
  int i = threadIdx.x;
  if (i >= NG * 5) return;
  ob[i * 2] = csb[i];
  ob[i * 2 + 1] = ceb[i];
}

__global__ void bn_stats(const float* __restrict__ r, float* bm, float* bv) {
  int t = threadIdx.x;
  if (t >= NG * 4) return;
  int b = t >> 2, c = t & 3;
  float s = 0.f, s2 = 0.f;
  for (int n = 0; n < NN; n++) {
    float v = r[((size_t)b * NN + n) * 4 + c];
    s += v; s2 += v * v;
  }
  float mu = s / NN;
  bm[t] = mu;
  bv[t] = s2 / NN - mu * mu;
}

__global__ void bn_apply(float* __restrict__ r, const float* __restrict__ bm,
                         const float* __restrict__ bv, const float* __restrict__ g,
                         const float* __restrict__ bb) {
  int idx = blockIdx.x * blockDim.x + threadIdx.x;
  if (idx >= NG * NN * 4) return;
  int c = idx & 3;
  int b = idx / (NN * 4);
  int t = b * 4 + c;
  float v = (r[idx] - bm[t]) * rsqrtf(bv[t] + 1e-5f) * g[c] + bb[c];
  r[idx] = fmaxf(v, 0.f);
}

__global__ void edge_pred(const float* __restrict__ ea, const int* __restrict__ src,
                          const int* __restrict__ dst, const float* __restrict__ r,
                          const float* __restrict__ ec_w, const float* __restrict__ ec_b,
                          float* __restrict__ out) {
  int idx = blockIdx.x * blockDim.x + threadIdx.x;
  if (idx >= NG * NE) return;
  int b = idx / NE;
  int s = src[idx], d = dst[idx];
  for (int j = 0; j < 2; j++) {
    float acc = ec_b[j];
    for (int k = 0; k < 16; k++) acc += ea[(size_t)idx * 16 + k] * ec_w[k * 2 + j];
    for (int k = 0; k < 4; k++)  acc += r[((size_t)b * NN + s) * 4 + k] * ec_w[(16 + k) * 2 + j];
    for (int k = 0; k < 4; k++)  acc += r[((size_t)b * NN + d) * 4 + k] * ec_w[(20 + k) * 2 + j];
    out[(size_t)idx * 2 + j] = acc;
  }
}

// --------------------------- host orchestration ----------------------------

enum {
  I_X = 0, I_EA, I_SRC, I_DST,
  I_G1_W, I_G1_WE, I_G1_AS, I_G1_AD, I_G1_AE, I_G1_B,
  I_G2_W, I_G2_WE, I_G2_AS, I_G2_AD, I_G2_AE, I_G2_B,
  I_TE_W, I_TE_B, I_IN_W, I_IN_B, I_OUT_W, I_OUT_B,
  I_IC_W1, I_IC_B1, I_IC_W2, I_IC_B2,
  I_R0_W1, I_R0_B1, I_R0_W2, I_R0_B2,
  I_R1_W1, I_R1_B1, I_R1_W2, I_R1_B2,
  I_R2_W1, I_R2_B1, I_R2_W2, I_R2_B2,
  I_WP, I_QP_W, I_QP_B,
  I_NC_W, I_NC_B, I_VAD_W, I_VAD_B, I_RED_W, I_RED_B,
  I_BN_G, I_BN_B, I_EC_W, I_EC_B
};

static inline void gemm(hipStream_t st, const float* A, const float* Bm, const float* bias,
                        float* C, int M, int N, int K, int transB, int relu) {
  dim3 g(CDIV(N, 32), CDIV(M, 32));
  gemm_wmma<<<g, 128, 0, st>>>(A, Bm, bias, C, M, N, K, transB, relu);
}

extern "C" void kernel_launch(void* const* d_in, const int* in_sizes, int n_in,
                              void* d_out, int out_size, void* d_ws, size_t ws_size,
                              hipStream_t stream) {
  const float* x    = (const float*)d_in[I_X];
  const float* ea   = (const float*)d_in[I_EA];
  const int*   src  = (const int*)d_in[I_SRC];
  const int*   dst  = (const int*)d_in[I_DST];
  float* out = (float*)d_out;

  // ---- workspace carve-out ----
  size_t cur = 0;
  auto WS = [&](size_t nfloats) -> float* {
    float* p = (float*)((char*)d_ws + cur);
    cur += ((nfloats * sizeof(float) + 255) / 256) * 256;
    return p;
  };
  float* deg    = WS((size_t)NG * NN);
  float* easum  = WS((size_t)NG * NN * 16);
  float* ea2    = WS((size_t)NG * NT * 16);
  float* hbuf   = WS((size_t)NG * NN * HGC);
  float* gacc   = WS((size_t)NG * NN * HGC);
  float* h1     = WS((size_t)NG * NN * CC);
  float* emb    = WS((size_t)NG * NN * CC);
  float* a_src  = WS((size_t)NG * NN * 2);
  float* a_dst  = WS((size_t)NG * NN * 2);
  float* a_e    = WS((size_t)NG * NT * 2);
  float* wbuf   = WS((size_t)NG * NT * 2);
  float* segm   = WS((size_t)NG * NN * 2);
  float* segs   = WS((size_t)NG * NN * 2);
  float* wae    = WS(32);
  float* xte    = WS((size_t)NG * NN * DD);
  float* qkv    = WS((size_t)NG * NN * 384);
  float* obuf   = WS((size_t)NG * NN * DD);
  float* ao     = WS((size_t)NG * NN * DD);
  float* kh2    = WS((size_t)NG * NN * DD);
  float* qbuf   = WS((size_t)NG * NN * 130);
  float* lg1    = WS((size_t)NG * NN * DD);
  float* lgb    = WS((size_t)NG * NN * 200);
  float* sob    = WS((size_t)NG * NN);
  float* eob    = WS((size_t)NG * NN);
  float* qp     = WS((size_t)NG * NN * DD);
  float* qph    = WS((size_t)NG * NN * DD);
  float* rsm    = WS((size_t)NG * HA * NN);
  float* rsl    = WS((size_t)NG * HA * NN);
  float* colsum = WS((size_t)NG * NN);
  float* csb    = WS(NG * 5);
  float* ceb    = WS(NG * 5);
  float* rbuf   = WS((size_t)NG * NN * 4);
  float* bnm    = WS(NG * 4);
  float* bnv    = WS(NG * 4);

  const int M = NG * NN;   // 8192 batched rows

  // ---- shared GAT edge statistics (same for both layers) ----
  hipMemsetAsync(deg,   0, (size_t)NG * NN * 4, stream);
  hipMemsetAsync(easum, 0, (size_t)NG * NN * 16 * 4, stream);
  edge_accum<<<CDIV(NG * NE, 256), 256, 0, stream>>>(ea, dst, deg, easum);
  build_ea2<<<CDIV(NG * NT * 16, 256), 256, 0, stream>>>(ea, easum, deg, ea2);

  // ---- GAT layer (shared routine) ----
  auto gat_layer = [&](const float* xin, int fin, int W_i, int WE_i, int AS_i,
                       int AD_i, int AE_i, int B_i, float* outp, int relu) {
    wae_kernel<<<1, 32, 0, stream>>>((const float*)d_in[WE_i], (const float*)d_in[AE_i], wae);
    ae_kernel<<<CDIV(NG * NT * 2, 256), 256, 0, stream>>>(ea2, wae, a_e);
    gemm(stream, xin, (const float*)d_in[W_i], nullptr, hbuf, M, HGC, fin, 0, 0);
    asrc_adst<<<CDIV(NG * NN * 2, 256), 256, 0, stream>>>(
        hbuf, (const float*)d_in[AS_i], (const float*)d_in[AD_i], a_src, a_dst);
    fill_f32<<<CDIV(NG * NN * 2, 256), 256, 0, stream>>>(segm, -1e30f, NG * NN * 2);
    hipMemsetAsync(segs, 0, (size_t)NG * NN * 2 * 4, stream);
    hipMemsetAsync(gacc, 0, (size_t)NG * NN * HGC * 4, stream);
    alpha_max<<<CDIV(NG * NT * 2, 256), 256, 0, stream>>>(a_src, a_dst, a_e, src, dst, wbuf, segm);
    expsum<<<CDIV(NG * NT * 2, 256), 256, 0, stream>>>(src, dst, segm, wbuf, segs);
    aggregate<<<CDIV(NG * NT * CC, 256), 256, 0, stream>>>(hbuf, src, dst, wbuf, segs, gacc);
    gat_final<<<CDIV(NG * NN * CC, 256), 256, 0, stream>>>(gacc, (const float*)d_in[B_i], outp, relu);
  };

  gat_layer(x,  FIN, I_G1_W, I_G1_WE, I_G1_AS, I_G1_AD, I_G1_AE, I_G1_B, h1,  1);
  gat_layer(h1, CC,  I_G2_W, I_G2_WE, I_G2_AS, I_G2_AD, I_G2_AE, I_G2_B, emb, 0);

  // ---- FDR ----
  add_te<<<CDIV(NG * NN * DD, 256), 256, 0, stream>>>(
      emb, (const float*)d_in[I_TE_W], (const float*)d_in[I_TE_B], xte);
  // qkv = x @ in_w.T + in_b   (in_w is [384,128] => NT mode)
  gemm(stream, xte, (const float*)d_in[I_IN_W], (const float*)d_in[I_IN_B],
       qkv, M, 384, DD, 1, 0);
  flash_mha<<<dim3(NN / 16, HA, NG), 32, 0, stream>>>(qkv, obuf);
  // ao = o @ out_w.T + out_b
  gemm(stream, obuf, (const float*)d_in[I_OUT_W], (const float*)d_in[I_OUT_B],
       ao, M, DD, DD, 1, 0);
  pooled_mask<<<NG, 128, 0, stream>>>(ao, (const float*)d_in[I_IC_W1],
      (const float*)d_in[I_IC_B1], (const float*)d_in[I_IC_W2],
      (const float*)d_in[I_IC_B2], out + O_MSK);
  // kh2 = ao @ wk.T + bk  (rows 128..255 of in_w)
  gemm(stream, ao, (const float*)d_in[I_IN_W] + 128 * 128,
       (const float*)d_in[I_IN_B] + 128, kh2, M, DD, DD, 1, 0);
  init_intervals<<<1, 64, 0, stream>>>(csb, ceb);

  for (int l = 0; l < 3; l++) {
    const float* w1 = (const float*)d_in[I_R0_W1 + l * 4];
    const float* b1 = (const float*)d_in[I_R0_B1 + l * 4];
    const float* w2 = (const float*)d_in[I_R0_W2 + l * 4];
    const float* b2 = (const float*)d_in[I_R0_B2 + l * 4];
    for (int ivx = 0; ivx < 5; ivx++) {
      build_q<<<CDIV(NG * NN * 130, 256), 256, 0, stream>>>(ao, csb, ceb, ivx, qbuf);
      gemm(stream, qbuf, w1, b1, lg1, M, DD, 130, 0, 1);          // relu(q@w1+b1)
      gemm(stream, lg1, w2, b2, lgb, M, 200, DD, 0, 0);
      soeo_kernel<<<CDIV(NG * NN, 256), 256, 0, stream>>>(lgb, (const float*)d_in[I_WP], sob, eob);
      gemm(stream, qbuf, (const float*)d_in[I_QP_W], (const float*)d_in[I_QP_B],
           qp, M, DD, 130, 1, 0);                                  // qp = q@qp_w.T+qp_b
      gemm(stream, qp, (const float*)d_in[I_IN_W], (const float*)d_in[I_IN_B],
           qph, M, DD, DD, 1, 0);                                  // qph = qp@wq.T+bq
      rowstats<<<CDIV(NG * HA * NN, 256), 256, 0, stream>>>(qph, kh2, rsm, rsl);
      hipMemsetAsync(colsum, 0, (size_t)NG * NN * 4, stream);
      colsum_acc<<<CDIV(NG * HA * NN, 256), 256, 0, stream>>>(qph, kh2, rsm, rsl, colsum);
      update_intervals<<<NG, 256, 0, stream>>>(colsum, sob, eob, csb, ceb, ivx);
    }
  }
  write_bounds<<<1, 64, 0, stream>>>(csb, ceb, out + O_BND);

  // ---- heads ----
  gemm(stream, emb, (const float*)d_in[I_NC_W], (const float*)d_in[I_NC_B],
       out + O_NP, M, 7, DD, 0, 0);
  gemm(stream, emb, (const float*)d_in[I_VAD_W], (const float*)d_in[I_VAD_B],
       out + O_VAD, M, 2, DD, 0, 0);
  gemm(stream, emb, (const float*)d_in[I_RED_W], (const float*)d_in[I_RED_B],
       rbuf, M, 4, DD, 0, 0);
  bn_stats<<<1, 32, 0, stream>>>(rbuf, bnm, bnv);
  bn_apply<<<CDIV(NG * NN * 4, 256), 256, 0, stream>>>(
      rbuf, bnm, bnv, (const float*)d_in[I_BN_G], (const float*)d_in[I_BN_B]);
  edge_pred<<<CDIV(NG * NE, 256), 256, 0, stream>>>(
      ea, src, dst, rbuf, (const float*)d_in[I_EC_W], (const float*)d_in[I_EC_B],
      out + O_EP);
}
